// MixAxialPOLABlock_21620865368136
// MI455X (gfx1250) — compile-verified, matrix operand order checked by reading, not run on hardware
//
#include <hip/hip_runtime.h>
#include <hip/hip_bf16.h>
#include <math.h>

typedef __attribute__((ext_vector_type(16))) _Float16     v16h;
typedef __attribute__((ext_vector_type(8)))  float        v8f;
typedef __attribute__((ext_vector_type(4)))  unsigned int v4u;
typedef __attribute__((ext_vector_type(8)))  int          v8i;
typedef __attribute__((ext_vector_type(4)))  int          v4i;

// TDM availability is a *device-pass* property: the host pass of hipcc never
// sees AMDGCN builtins, so only probe (and only hard-fail) when compiling
// for the device.  Host pass parses the manual-staging fallback instead.
#if defined(__HIP_DEVICE_COMPILE__) && defined(__has_builtin)
#if __has_builtin(__builtin_amdgcn_tensor_load_to_lds)
#define HAVE_TDM 1
#endif
#endif
#ifndef HAVE_TDM
#define HAVE_TDM 0
#endif

#if defined(__HIP_DEVICE_COMPILE__) && !HAVE_TDM
#error "device pass: gfx1250 tensor_load_to_lds builtin not available"
#endif

// ---------------------------------------------------------------------------
// WMMA helpers (CDNA5 gfx1250, wave32, 16x16x32 f16 -> f32)
//
// A-frag (16x32, row-major in LDS, stride `stride` halfs):
//   lane<16 : M=lane,    K in {0..7, 16..23}
//   lane>=16: M=lane-16, K in {8..15,24..31}
// B-frag is identical if the B matrix is staged *transposed* (N-major rows).
// C/D frag: reg r, lane l -> (M = r + 8*(l>>4), N = l&15).
// ---------------------------------------------------------------------------
static __device__ __forceinline__ v16h ldfrag(const _Float16* base, int stride,
                                              int r0, int lane) {
  int m  = r0 + (lane & 15);
  int k0 = (lane >> 4) << 3;
  const _Float16* p = base + m * stride + k0;
  v16h f;
  ((uint4*)&f)[0] = *(const uint4*)(p);
  ((uint4*)&f)[1] = *(const uint4*)(p + 16);
  return f;
}

static __device__ __forceinline__ v8f wmma32(v16h a, v16h b, v8f c) {
  return __builtin_amdgcn_wmma_f32_16x16x32_f16(false, a, false, b,
                                                (short)0, c, false, false);
}

// ---------------------------------------------------------------------------
// Tensor Data Mover: DMA a (rows x 32) f16 tile (row stride lda elements)
// from global memory into LDS.  2D descriptor per CDNA5 ISA 8.3/8.4:
//   group0: count=1 | lds_addr | global_addr | type=2
//   group1: data_size=1(2B), tensor_dim0=32, tensor_dim1=tensor_rows,
//           tile_dim0=32, tile_dim1=rows, tensor_dim0_stride=lda
// Rows past tensor_dim1 read as zero (tail of M handled for free).
// ---------------------------------------------------------------------------
#if HAVE_TDM
static __device__ __forceinline__ void tdm_load_a32(const _Float16* gptr,
                                                    _Float16* lptr, int rows,
                                                    int lda, int tensor_rows) {
  unsigned long long ga = (unsigned long long)(uintptr_t)gptr;
  unsigned int lds = (unsigned int)(uintptr_t)lptr;
  v4u g0;
  g0[0] = 1u;                                             // count = 1
  g0[1] = lds;                                            // lds_addr
  g0[2] = (unsigned int)(ga & 0xFFFFFFFFu);               // global_addr lo
  g0[3] = (unsigned int)((ga >> 32) & 0x01FFFFFFu) | (2u << 30); // hi | type=2
  unsigned int td0 = 32u, tile0 = 32u;
  unsigned int td1 = (unsigned int)tensor_rows;
  v8i g1;
  g1[0] = (int)0x00010000u;                               // data_size = 2B
  g1[1] = (int)((td0 & 0xFFFFu) << 16);                   // tensor_dim0 lo16
  g1[2] = (int)((td0 >> 16) | ((td1 & 0xFFFFu) << 16));   // td0 hi | td1 lo
  g1[3] = (int)((td1 >> 16) | (tile0 << 16));             // td1 hi | tile_dim0
  g1[4] = (int)(rows & 0xFFFF);                           // tile_dim1 (dim2=0)
  g1[5] = (int)(unsigned int)lda;                         // dim0_stride lo32
  g1[6] = 0;
  g1[7] = 0;
  v4i z4 = {0, 0, 0, 0};
#if __clang_major__ >= 23
  v8i z8 = {0, 0, 0, 0, 0, 0, 0, 0};
  __builtin_amdgcn_tensor_load_to_lds(g0, g1, z4, z4, z8, 0);
#else
  __builtin_amdgcn_tensor_load_to_lds(g0, g1, z4, z4, 0);
#endif
}
#endif

// ---------------------------------------------------------------------------
// LayerNorm over C=256, one wave per token, f32 in -> f16 out
// ---------------------------------------------------------------------------
__global__ __launch_bounds__(256) void ln256_kernel(
    const float* __restrict__ x, const float* __restrict__ w,
    const float* __restrict__ b, _Float16* __restrict__ out, int ntok) {
  int wv = threadIdx.x >> 5, lane = threadIdx.x & 31;
  int tok = blockIdx.x * 8 + wv;
  if (tok >= ntok) return;
  const float* xr = x + (size_t)tok * 256 + lane * 8;
  float v[8];
  float4 a0 = *(const float4*)(xr);
  float4 a1 = *(const float4*)(xr + 4);
  v[0]=a0.x; v[1]=a0.y; v[2]=a0.z; v[3]=a0.w;
  v[4]=a1.x; v[5]=a1.y; v[6]=a1.z; v[7]=a1.w;
  float s = 0.f, sq = 0.f;
#pragma unroll
  for (int i = 0; i < 8; ++i) { s += v[i]; sq += v[i]*v[i]; }
#pragma unroll
  for (int m = 16; m >= 1; m >>= 1) {
    s  += __shfl_xor(s,  m, 32);
    sq += __shfl_xor(sq, m, 32);
  }
  float mu  = s * (1.f/256.f);
  float var = sq * (1.f/256.f) - mu*mu;
  float rs  = rsqrtf(var + 1e-5f);
  _Float16* o = out + (size_t)tok * 256 + lane * 8;
#pragma unroll
  for (int i = 0; i < 8; ++i) {
    int c = lane * 8 + i;
    o[i] = (_Float16)((v[i] - mu) * rs * w[c] + b[c]);
  }
}

// ---------------------------------------------------------------------------
// Build zero-padded local-channel map: (B,112,112,:128) -> (B,126,126,128)
// ---------------------------------------------------------------------------
__global__ __launch_bounds__(256) void pad_local_kernel(
    const _Float16* __restrict__ h1, _Float16* __restrict__ xlp) {
  int idx = blockIdx.x * blockDim.x + threadIdx.x;
  const int total = 4 * 126 * 126 * 8;
  if (idx >= total) return;
  int part = (idx & 7) * 16;
  int row  = idx >> 3;
  int pw = row % 126;
  int t  = row / 126;
  int ph = t % 126;
  int b  = t / 126;
  _Float16* dst = xlp + (size_t)row * 128 + part;
  if (ph >= 7 && ph < 119 && pw >= 7 && pw < 119) {
    const _Float16* src =
        h1 + ((size_t)((b*112 + (ph-7))*112 + (pw-7)))*256 + part;
    ((uint4*)dst)[0] = ((const uint4*)src)[0];
    ((uint4*)dst)[1] = ((const uint4*)src)[1];
  } else {
    uint4 z = {0,0,0,0};
    ((uint4*)dst)[0] = z;
    ((uint4*)dst)[1] = z;
  }
}

// ---------------------------------------------------------------------------
// Expand the (729,4) rel-pos bias table to a dense (4,49,448) map.
// Columns 441..447 (K padding) get -1e30 so softmax masking is free.
// Done once; the hot attention loop then needs zero div/mod arithmetic.
// ---------------------------------------------------------------------------
__global__ __launch_bounds__(256) void expand_bias_kernel(
    const float* __restrict__ btab, float* __restrict__ bmap) {
  int idx = blockIdx.x * blockDim.x + threadIdx.x;
  const int total = 4 * 49 * 448;
  if (idx >= total) return;
  int k = idx % 448;
  int t = idx / 448;
  int q = t % 49;
  int head = t / 49;
  float v = -1e30f;
  if (k < 441) {
    int qi = q / 7, qj = q % 7;
    int ki = k / 21, kj = k % 21;
    v = btab[((qi - ki + 20) * 27 + (qj - kj + 20)) * 4 + head];
  }
  bmap[idx] = v;
}

// ---------------------------------------------------------------------------
// WMMA GEMM, 64x64 block tile (for N=64 matmuls).
// 4 waves, each 32x32 (2x2 WMMA frags).  Prefetches next A k-tile.
// ---------------------------------------------------------------------------
template<int ACT, int OUTF32>
__global__ __launch_bounds__(128) void gemm_kernel(
    const _Float16* __restrict__ A, int lda, int M,
    const float* __restrict__ Bw, const float* __restrict__ bias,
    int N, int K,
    void* __restrict__ Out, int ldo, int co,
    const float* __restrict__ resid, int ldr) {
  __shared__ __align__(16) _Float16 sA[64 * 32];
  __shared__ __align__(16) _Float16 sB[64 * 32];   // transposed: [n][k]
  int tid = threadIdx.x, lane = tid & 31, wv = tid >> 5;
  int m0blk = blockIdx.x * 64, n0blk = blockIdx.y * 64;
  int wm = (wv >> 1) * 32, wn = (wv & 1) * 32;
  v8f acc[2][2];
#pragma unroll
  for (int i = 0; i < 2; ++i)
#pragma unroll
    for (int j = 0; j < 2; ++j) acc[i][j] = (v8f){};

  for (int k0 = 0; k0 < K; k0 += 32) {
    __syncthreads();
#pragma unroll
    for (int i = 0; i < 2; ++i) {          // stage A tile (64x32 f16)
      int u = i * 128 + tid;
      int row = u >> 2, part = (u & 3) * 8;
      int gr = m0blk + row; if (gr >= M) gr = M - 1;
      *(uint4*)(sA + row*32 + part) =
          *(const uint4*)(A + (size_t)gr * lda + k0 + part);
    }
    if (k0 + 32 < K) {                     // prefetch next A k-tile
      int pr = m0blk + (tid & 63); if (pr >= M) pr = M - 1;
      __builtin_prefetch(A + (size_t)pr * lda + k0 + 32, 0, 1);
    }
#pragma unroll
    for (int i = 0; i < 16; ++i) {         // stage W tile transposed, f32->f16
      int e = i * 128 + tid;
      int kk = e >> 6, nn = e & 63;
      sB[nn*32 + kk] = (_Float16)Bw[(size_t)(k0 + kk) * N + n0blk + nn];
    }
    __syncthreads();
    v16h af0 = ldfrag(sA, 32, wm,      lane);
    v16h af1 = ldfrag(sA, 32, wm + 16, lane);
    v16h bf0 = ldfrag(sB, 32, wn,      lane);
    v16h bf1 = ldfrag(sB, 32, wn + 16, lane);
    acc[0][0] = wmma32(af0, bf0, acc[0][0]);
    acc[0][1] = wmma32(af0, bf1, acc[0][1]);
    acc[1][0] = wmma32(af1, bf0, acc[1][0]);
    acc[1][1] = wmma32(af1, bf1, acc[1][1]);
  }

  int colLo = lane & 15, hi = lane >> 4;
#pragma unroll
  for (int sm = 0; sm < 2; ++sm)
#pragma unroll
    for (int sn = 0; sn < 2; ++sn)
#pragma unroll
      for (int r = 0; r < 8; ++r) {
        int row = m0blk + wm + sm*16 + r + 8*hi;
        int col = n0blk + wn + sn*16 + colLo;
        if (row < M) {
          float v = acc[sm][sn][r] + bias[col];
          if (ACT == 1) v = 0.5f * v * (1.f + erff(v * 0.70710678f));
          if (OUTF32) {
            float* O = (float*)Out;
            size_t o = (size_t)row * ldo + co + col;
            float rr = resid ? resid[(size_t)row * ldr + co + col] : 0.f;
            O[o] = rr + v;
          } else {
            ((_Float16*)Out)[(size_t)row * ldo + col] = (_Float16)v;
          }
        }
      }
}

// ---------------------------------------------------------------------------
// WMMA GEMM, 128x128 block tile (N multiple of 128).
// 256 threads / 8 waves (4x2); wave computes 32x64 = 2x4 frags, 8 WMMA/k-step.
// Double-buffered LDS: Tensor Data Mover DMAs A tile k+1 while WMMAs consume
// tile k; s_wait_tensorcnt only at the iteration boundary.
// ---------------------------------------------------------------------------
template<int ACT, int OUTF32>
__global__ __launch_bounds__(256) void gemm128_kernel(
    const _Float16* __restrict__ A, int lda, int M,
    const float* __restrict__ Bw, const float* __restrict__ bias,
    int N, int K,
    void* __restrict__ Out, int ldo, int co,
    const float* __restrict__ resid, int ldr) {
  __shared__ __align__(16) _Float16 sA[2][128 * 32];
  __shared__ __align__(16) _Float16 sB[2][128 * 32];   // transposed: [n][k]
  int tid = threadIdx.x, lane = tid & 31, wv = tid >> 5;
  int m0blk = blockIdx.x * 128, n0blk = blockIdx.y * 128;
  int wm = (wv >> 1) * 32, wn = (wv & 1) * 64;
  v8f acc[2][4];
#pragma unroll
  for (int i = 0; i < 2; ++i)
#pragma unroll
    for (int j = 0; j < 4; ++j) acc[i][j] = (v8f){};

  auto stageA = [&](int k0s, int buf) {
#if HAVE_TDM
    if (wv == 0)
      tdm_load_a32(A + (size_t)m0blk * lda + k0s, sA[buf], 128, lda, M - m0blk);
#else
#pragma unroll
    for (int i = 0; i < 2; ++i) {
      int u = i * 256 + tid;
      int row = u >> 2, part = (u & 3) * 8;
      int gr = m0blk + row; if (gr >= M) gr = M - 1;
      *(uint4*)(sA[buf] + row*32 + part) =
          *(const uint4*)(A + (size_t)gr * lda + k0s + part);
    }
#endif
  };
  auto stageB = [&](int k0s, int buf) {
#pragma unroll
    for (int i = 0; i < 16; ++i) {
      int e = i * 256 + tid;
      int kk = e >> 7, nn = e & 127;
      sB[buf][nn*32 + kk] = (_Float16)Bw[(size_t)(k0s + kk) * N + n0blk + nn];
    }
  };

  const int nk = K >> 5;
  stageA(0, 0);                            // prologue: tile 0 -> buffer 0
  stageB(0, 0);
#if HAVE_TDM
  __builtin_amdgcn_s_wait_tensorcnt(0);
#endif

  for (int k = 0; k < nk; ++k) {
    __syncthreads();                       // publish buffer k&1
    int cur = k & 1, nxt = cur ^ 1;
    if (k + 1 < nk) {                      // stage tile k+1 while computing k
      stageA((k + 1) * 32, nxt);
      stageB((k + 1) * 32, nxt);
    }
    v16h af0 = ldfrag(sA[cur], 32, wm,      lane);
    v16h af1 = ldfrag(sA[cur], 32, wm + 16, lane);
    v16h bf[4];
#pragma unroll
    for (int j = 0; j < 4; ++j) bf[j] = ldfrag(sB[cur], 32, wn + j*16, lane);
#pragma unroll
    for (int j = 0; j < 4; ++j) {
      acc[0][j] = wmma32(af0, bf[j], acc[0][j]);
      acc[1][j] = wmma32(af1, bf[j], acc[1][j]);
    }
#if HAVE_TDM
    __builtin_amdgcn_s_wait_tensorcnt(0);  // A(k+1) DMA done before publish
#endif
  }

  int colLo = lane & 15, hi = lane >> 4;
#pragma unroll
  for (int sm = 0; sm < 2; ++sm)
#pragma unroll
    for (int sn = 0; sn < 4; ++sn)
#pragma unroll
      for (int r = 0; r < 8; ++r) {
        int row = m0blk + wm + sm*16 + r + 8*hi;
        int col = n0blk + wn + sn*16 + colLo;
        if (row < M) {
          float v = acc[sm][sn][r] + bias[col];
          if (ACT == 1) v = 0.5f * v * (1.f + erff(v * 0.70710678f));
          if (OUTF32) {
            float* O = (float*)Out;
            size_t o = (size_t)row * ldo + co + col;
            float rr = resid ? resid[(size_t)row * ldr + co + col] : 0.f;
            O[o] = rr + v;
          } else {
            ((_Float16*)Out)[(size_t)row * ldo + col] = (_Float16)v;
          }
        }
      }
}

// ---------------------------------------------------------------------------
// Local 7x7-window attention over 21x21 padded KV neighborhood.
// One block = one (window, head); 4 waves, wave w owns query rows 16w..16w+15.
// Online (flash) softmax over 14 K-chunks of 32; bias via precomputed map
// (masking for k>=441 baked in as -1e30).
// ---------------------------------------------------------------------------
__global__ __launch_bounds__(128) void local_attn_kernel(
    const _Float16* __restrict__ Q,    // (B*112*112, 128)
    const _Float16* __restrict__ Kp,   // (B*126*126, 128)
    const _Float16* __restrict__ Vp,   // (B*126*126, 128)
    const float*    __restrict__ bmap, // (4, 49, 448) expanded bias
    _Float16*       __restrict__ ctx)  // (B*112*112, 128)
{
  const int head = blockIdx.y;
  int win = blockIdx.x;
  int b = win >> 8, wh = (win >> 4) & 15, ww = win & 15;

  __shared__ __align__(16) _Float16 sQ[64 * 32];
  __shared__ __align__(16) _Float16 sK[32 * 32];
  __shared__ __align__(16) _Float16 sVt[32 * 32];  // [chan][k]
  __shared__ __align__(16) _Float16 sP[4 * 16 * 32];

  int tid = threadIdx.x, lane = tid & 31, wv = tid >> 5;
  int colLo = lane & 15, hi = lane >> 4;

  { // stage Q scaled by d^-0.5
    int r = tid >> 1, part = (tid & 1) * 16;
    if (r < 49) {
      int qi = r / 7, qj = r % 7;
      const _Float16* src =
          Q + ((size_t)((b*112 + wh*7+qi)*112 + ww*7+qj))*128 + head*32 + part;
#pragma unroll
      for (int i = 0; i < 16; ++i)
        sQ[r*32 + part + i] = (_Float16)((float)src[i] * 0.1767766953f);
    } else {
#pragma unroll
      for (int i = 0; i < 16; ++i) sQ[r*32 + part + i] = (_Float16)0.f;
    }
  }
  __syncthreads();

  v16h qf = ldfrag(sQ, 32, wv * 16, lane);
  v8f acc0 = (v8f){}, acc1 = (v8f){};
  float mrow[8], lrow[8];
#pragma unroll
  for (int r = 0; r < 8; ++r) { mrow[r] = -1e30f; lrow[r] = 0.f; }

  // per-register bias row pointers (q clamped; rows q>=49 are never stored)
  const float* brow[8];
#pragma unroll
  for (int r = 0; r < 8; ++r) {
    int q = wv*16 + r + 8*hi;
    int qq = q < 49 ? q : 48;
    brow[r] = bmap + (size_t)(head * 49 + qq) * 448;
  }

  for (int kc = 0; kc < 14; ++kc) {
    __syncthreads();
    if (tid < 64) {            // stage K chunk (32 rows x 32 chans, row-major)
      int r = tid >> 1, part = (tid & 1) * 16;
      int kidx = kc * 32 + r;
      if (kidx < 441) {
        int ki = kidx / 21, kj = kidx % 21;
        const _Float16* src =
            Kp + ((size_t)((b*126 + wh*7+ki)*126 + ww*7+kj))*128 + head*32 + part;
        *(uint4*)(sK + r*32 + part)     = ((const uint4*)src)[0];
        *(uint4*)(sK + r*32 + part + 8) = ((const uint4*)src)[1];
      } else {
        uint4 z = {0,0,0,0};
        *(uint4*)(sK + r*32 + part)     = z;
        *(uint4*)(sK + r*32 + part + 8) = z;
      }
    } else {                   // stage V chunk transposed [chan][k]
      int u = tid - 64;
#pragma unroll
      for (int i = 0; i < 16; ++i) {
        int e = u * 16 + i;
        int c = e >> 5, kk = e & 31;
        int kidx = kc * 32 + kk;
        _Float16 val = (_Float16)0.f;
        if (kidx < 441) {
          int ki = kidx / 21, kj = kidx % 21;
          val = Vp[((size_t)((b*126 + wh*7+ki)*126 + ww*7+kj))*128 + head*32 + c];
        }
        sVt[c*32 + kk] = val;
      }
    }
    __syncthreads();

    v16h bk0 = ldfrag(sK, 32, 0,  lane);
    v16h bk1 = ldfrag(sK, 32, 16, lane);
    v8f z8 = (v8f){};
    v8f s0 = wmma32(qf, bk0, z8);
    v8f s1 = wmma32(qf, bk1, z8);

#pragma unroll
    for (int r = 0; r < 8; ++r) {
      float v0 = s0[r] + brow[r][kc*32 + colLo];
      float v1 = s1[r] + brow[r][kc*32 + colLo + 16];

      float cmax = fmaxf(v0, v1);
      cmax = fmaxf(cmax, __shfl_xor(cmax, 1, 32));
      cmax = fmaxf(cmax, __shfl_xor(cmax, 2, 32));
      cmax = fmaxf(cmax, __shfl_xor(cmax, 4, 32));
      cmax = fmaxf(cmax, __shfl_xor(cmax, 8, 32));
      float nm = fmaxf(mrow[r], cmax);
      float sc = __expf(mrow[r] - nm);
      float p0 = __expf(v0 - nm), p1 = __expf(v1 - nm);
      float rsum = p0 + p1;
      rsum += __shfl_xor(rsum, 1, 32);
      rsum += __shfl_xor(rsum, 2, 32);
      rsum += __shfl_xor(rsum, 4, 32);
      rsum += __shfl_xor(rsum, 8, 32);
      lrow[r] = lrow[r] * sc + rsum;
      mrow[r] = nm;
      acc0[r] *= sc; acc1[r] *= sc;
      int m = r + 8*hi;                      // C-layout -> LDS for transpose
      sP[wv*512 + m*32 + colLo]      = (_Float16)p0;
      sP[wv*512 + m*32 + colLo + 16] = (_Float16)p1;
    }
    __syncthreads();
    v16h pf  = ldfrag(sP + wv*512, 32, 0,  lane);
    v16h bv0 = ldfrag(sVt, 32, 0,  lane);
    v16h bv1 = ldfrag(sVt, 32, 16, lane);
    acc0 = wmma32(pf, bv0, acc0);
    acc1 = wmma32(pf, bv1, acc1);
  }

#pragma unroll
  for (int r = 0; r < 8; ++r) {
    int q = wv*16 + r + 8*hi;
    if (q < 49) {
      float inv = 1.f / lrow[r];
      int qi = q / 7, qj = q % 7;
      size_t tok = (size_t)((b*112 + wh*7+qi)*112 + ww*7+qj);
      ctx[tok*128 + head*32 + colLo]      = (_Float16)(acc0[r] * inv);
      ctx[tok*128 + head*32 + colLo + 16] = (_Float16)(acc1[r] * inv);
    }
  }
}

// ---------------------------------------------------------------------------
// Axial attention: one block per (sequence bn, head); seq len 112, d=32.
// vertical=1: sequence over H (stride 112*64); vertical=0: over W (stride 64).
// ---------------------------------------------------------------------------
__global__ __launch_bounds__(128) void axial_attn_kernel(
    const _Float16* __restrict__ Qb, const _Float16* __restrict__ Kb,
    const _Float16* __restrict__ Vb, _Float16* __restrict__ outb,
    int vertical) {
  int bn = blockIdx.x, head = blockIdx.y;
  int b = bn / 112, o = bn % 112;
  size_t base; int stride;
  if (vertical) { base = ((size_t)b*12544 + o)        * 64; stride = 112*64; }
  else          { base = ((size_t)b*12544 + (size_t)o*112) * 64; stride = 64; }
  base += head * 32;

  __shared__ __align__(16) _Float16 sQ[128 * 32];
  __shared__ __align__(16) _Float16 sK[128 * 32];
  __shared__ __align__(16) _Float16 sVt[32 * 128];   // [chan][k]
  __shared__ __align__(16) _Float16 sP[4 * 16 * 128];

  int tid = threadIdx.x, lane = tid & 31, wv = tid >> 5;
  int colLo = lane & 15, hi = lane >> 4;

  for (int u = tid; u < 256; u += 128) {   // stage Q (scaled) and K
    int r = u >> 1, part = (u & 1) * 16;
    if (r < 112) {
      const _Float16* sq = Qb + base + (size_t)r * stride + part;
#pragma unroll
      for (int i = 0; i < 16; ++i)
        sQ[r*32 + part + i] = (_Float16)((float)sq[i] * 0.1767766953f);
      const uint4* sk = (const uint4*)(Kb + base + (size_t)r * stride + part);
      *(uint4*)(sK + r*32 + part)     = sk[0];
      *(uint4*)(sK + r*32 + part + 8) = sk[1];
    } else {
      uint4 z = {0,0,0,0};
#pragma unroll
      for (int i = 0; i < 16; ++i) sQ[r*32 + part + i] = (_Float16)0.f;
      *(uint4*)(sK + r*32 + part)     = z;
      *(uint4*)(sK + r*32 + part + 8) = z;
    }
  }
#pragma unroll
  for (int i = 0; i < 32; ++i) {           // stage V transposed
    int e = i * 128 + tid;
    int c = e >> 7, kk = e & 127;
    _Float16 val = (_Float16)0.f;
    if (kk < 112) val = Vb[base + (size_t)kk * stride + c];
    sVt[c*128 + kk] = val;
  }
  __syncthreads();

  for (int t = wv; t < 8; t += 4) {        // each wave: 2 row tiles of 16
    v16h qf = ldfrag(sQ, 32, t * 16, lane);
    v8f sf[8];
    v8f z8 = (v8f){};
#pragma unroll
    for (int j = 0; j < 8; ++j)
      sf[j] = wmma32(qf, ldfrag(sK, 32, j * 16, lane), z8);

#pragma unroll
    for (int r = 0; r < 8; ++r) {
      float mx = -1e30f;
#pragma unroll
      for (int j = 0; j < 8; ++j) {
        float v = sf[j][r];
        if (j*16 + colLo >= 112) v = -1e30f;
        sf[j][r] = v;
        mx = fmaxf(mx, v);
      }
      mx = fmaxf(mx, __shfl_xor(mx, 1, 32));
      mx = fmaxf(mx, __shfl_xor(mx, 2, 32));
      mx = fmaxf(mx, __shfl_xor(mx, 4, 32));
      mx = fmaxf(mx, __shfl_xor(mx, 8, 32));
      float sum = 0.f;
#pragma unroll
      for (int j = 0; j < 8; ++j) {
        float p = __expf(sf[j][r] - mx);
        sum += p;
        sf[j][r] = p;
      }
      sum += __shfl_xor(sum, 1, 32);
      sum += __shfl_xor(sum, 2, 32);
      sum += __shfl_xor(sum, 4, 32);
      sum += __shfl_xor(sum, 8, 32);
      float inv = 1.f / sum;
      int m = r + 8*hi;
#pragma unroll
      for (int j = 0; j < 8; ++j)
        sP[wv*2048 + m*128 + j*16 + colLo] = (_Float16)(sf[j][r] * inv);
    }
    __syncthreads();

    v8f a0 = (v8f){}, a1 = (v8f){};
#pragma unroll
    for (int kc = 0; kc < 4; ++kc) {
      v16h pf = ldfrag(sP + wv*2048 + kc*32, 128, 0,  lane);
      v16h b0 = ldfrag(sVt + kc*32,          128, 0,  lane);
      v16h b1 = ldfrag(sVt + kc*32,          128, 16, lane);
      a0 = wmma32(pf, b0, a0);
      a1 = wmma32(pf, b1, a1);
    }
#pragma unroll
    for (int r = 0; r < 8; ++r) {
      int i = t*16 + r + 8*hi;
      if (i < 112) {
        outb[base + (size_t)i * stride + colLo]      = (_Float16)a0[r];
        outb[base + (size_t)i * stride + colLo + 16] = (_Float16)a1[r];
      }
    }
  }
}

// ---------------------------------------------------------------------------
// Host-side orchestration
// ---------------------------------------------------------------------------
extern "C" void kernel_launch(void* const* d_in, const int* in_sizes, int n_in,
                              void* d_out, int out_size, void* d_ws, size_t ws_size,
                              hipStream_t stream) {
  const int Mtok = 4 * 112 * 112;     // 50176
  const int Mpad = 4 * 126 * 126;     // 63504

  const float* x       = (const float*)d_in[0];
  const float* ln1_w   = (const float*)d_in[3];
  const float* ln1_b   = (const float*)d_in[4];
  const float* ln2_w   = (const float*)d_in[5];
  const float* ln2_b   = (const float*)d_in[6];
  const float* btab    = (const float*)d_in[7];
  const float* loc_wq  = (const float*)d_in[8];
  const float* loc_bq  = (const float*)d_in[9];
  const float* loc_wk  = (const float*)d_in[10];
  const float* loc_bk  = (const float*)d_in[11];
  const float* loc_wv  = (const float*)d_in[12];
  const float* loc_bv  = (const float*)d_in[13];
  const float* loc_wp  = (const float*)d_in[14];
  const float* loc_bp  = (const float*)d_in[15];
  const float* v_wq    = (const float*)d_in[16];
  const float* v_bq    = (const float*)d_in[17];
  const float* v_wk    = (const float*)d_in[18];
  const float* v_bk    = (const float*)d_in[19];
  const float* v_wv    = (const float*)d_in[20];
  const float* v_bv    = (const float*)d_in[21];
  const float* v_wp    = (const float*)d_in[22];
  const float* v_bp    = (const float*)d_in[23];
  const float* h_wq    = (const float*)d_in[24];
  const float* h_bq    = (const float*)d_in[25];
  const float* h_wk    = (const float*)d_in[26];
  const float* h_bk    = (const float*)d_in[27];
  const float* h_wv    = (const float*)d_in[28];
  const float* h_bv    = (const float*)d_in[29];
  const float* h_wp    = (const float*)d_in[30];
  const float* h_bp    = (const float*)d_in[31];
  const float* fc1_w   = (const float*)d_in[32];
  const float* fc1_b   = (const float*)d_in[33];
  const float* fc2_w   = (const float*)d_in[34];
  const float* fc2_b   = (const float*)d_in[35];
  float* out = (float*)d_out;

  unsigned char* wsp = (unsigned char*)d_ws;
  size_t off = 0;
  auto alloc = [&](size_t bytes) -> void* {
    void* p = wsp + off;
    off += (bytes + 255) & ~(size_t)255;
    return p;
  };
  _Float16* h1     = (_Float16*)alloc((size_t)Mtok * 256 * 2);
  _Float16* xlp    = (_Float16*)alloc((size_t)Mpad * 128 * 2);
  _Float16* kp     = (_Float16*)alloc((size_t)Mpad * 128 * 2);
  _Float16* vp     = (_Float16*)alloc((size_t)Mpad * 128 * 2);
  _Float16* qloc   = (_Float16*)alloc((size_t)Mtok * 128 * 2);
  _Float16* ctxloc = (_Float16*)alloc((size_t)Mtok * 128 * 2);
  _Float16* qv     = (_Float16*)alloc((size_t)Mtok * 64 * 2);
  _Float16* kv     = (_Float16*)alloc((size_t)Mtok * 64 * 2);
  _Float16* vv     = (_Float16*)alloc((size_t)Mtok * 64 * 2);
  _Float16* qh     = (_Float16*)alloc((size_t)Mtok * 64 * 2);
  _Float16* kh     = (_Float16*)alloc((size_t)Mtok * 64 * 2);
  _Float16* vh     = (_Float16*)alloc((size_t)Mtok * 64 * 2);
  _Float16* ctxv   = (_Float16*)alloc((size_t)Mtok * 64 * 2);
  _Float16* ctxh   = (_Float16*)alloc((size_t)Mtok * 64 * 2);
  float*    x2     = (float*)   alloc((size_t)Mtok * 256 * 4);
  _Float16* h2a    = (_Float16*)alloc((size_t)Mtok * 256 * 2);
  _Float16* h2b    = (_Float16*)alloc((size_t)Mtok * 1024 * 2);
  float*    bmap   = (float*)   alloc((size_t)4 * 49 * 448 * 4);

  // 0) expand rel-pos bias table to dense masked map
  expand_bias_kernel<<<(4*49*448 + 255) / 256, 256, 0, stream>>>(btab, bmap);

  // 1) LN1 -> h1 (f16)
  ln256_kernel<<<(Mtok + 7) / 8, 256, 0, stream>>>(x, ln1_w, ln1_b, h1, Mtok);

  // 2) zero-padded local map
  {
    int tot = Mpad * 8;
    pad_local_kernel<<<(tot + 255) / 256, 256, 0, stream>>>(h1, xlp);
  }

  // 3) local Q / K / V projections (K,V on padded map: pad rows -> bias)
  gemm128_kernel<0,0><<<dim3((Mtok+127)/128, 1), 256, 0, stream>>>(
      h1, 256, Mtok, loc_wq, loc_bq, 128, 128, qloc, 128, 0, nullptr, 0);
  gemm128_kernel<0,0><<<dim3((Mpad+127)/128, 1), 256, 0, stream>>>(
      xlp, 128, Mpad, loc_wk, loc_bk, 128, 128, kp, 128, 0, nullptr, 0);
  gemm128_kernel<0,0><<<dim3((Mpad+127)/128, 1), 256, 0, stream>>>(
      xlp, 128, Mpad, loc_wv, loc_bv, 128, 128, vp, 128, 0, nullptr, 0);

  // 4) local window attention
  local_attn_kernel<<<dim3(1024, 4), 128, 0, stream>>>(qloc, kp, vp, bmap, ctxloc);

  // 5) local output projection -> x2[:, 0:128] (+ residual x)
  gemm128_kernel<0,1><<<dim3((Mtok+127)/128, 1), 256, 0, stream>>>(
      ctxloc, 128, Mtok, loc_wp, loc_bp, 128, 128, x2, 256, 0, x, 256);

  // 6) axial projections (vertical uses channels 128..191, horiz 192..255)
  gemm_kernel<0,0><<<dim3((Mtok+63)/64, 1), 128, 0, stream>>>(
      h1 + 128, 256, Mtok, v_wq, v_bq, 64, 64, qv, 64, 0, nullptr, 0);
  gemm_kernel<0,0><<<dim3((Mtok+63)/64, 1), 128, 0, stream>>>(
      h1 + 128, 256, Mtok, v_wk, v_bk, 64, 64, kv, 64, 0, nullptr, 0);
  gemm_kernel<0,0><<<dim3((Mtok+63)/64, 1), 128, 0, stream>>>(
      h1 + 128, 256, Mtok, v_wv, v_bv, 64, 64, vv, 64, 0, nullptr, 0);
  gemm_kernel<0,0><<<dim3((Mtok+63)/64, 1), 128, 0, stream>>>(
      h1 + 192, 256, Mtok, h_wq, h_bq, 64, 64, qh, 64, 0, nullptr, 0);
  gemm_kernel<0,0><<<dim3((Mtok+63)/64, 1), 128, 0, stream>>>(
      h1 + 192, 256, Mtok, h_wk, h_bk, 64, 64, kh, 64, 0, nullptr, 0);
  gemm_kernel<0,0><<<dim3((Mtok+63)/64, 1), 128, 0, stream>>>(
      h1 + 192, 256, Mtok, h_wv, h_bv, 64, 64, vh, 64, 0, nullptr, 0);

  // 7) axial attentions
  axial_attn_kernel<<<dim3(448, 2), 128, 0, stream>>>(qv, kv, vv, ctxv, 1);
  axial_attn_kernel<<<dim3(448, 2), 128, 0, stream>>>(qh, kh, vh, ctxh, 0);

  // 8) axial output projections -> x2[:,128:192], x2[:,192:256] (+ residual x)
  gemm_kernel<0,1><<<dim3((Mtok+63)/64, 1), 128, 0, stream>>>(
      ctxv, 64, Mtok, v_wp, v_bp, 64, 64, x2, 256, 128, x, 256);
  gemm_kernel<0,1><<<dim3((Mtok+63)/64, 1), 128, 0, stream>>>(
      ctxh, 64, Mtok, h_wp, h_bp, 64, 64, x2, 256, 192, x, 256);

  // 9) LN2 -> h2a (f16)
  ln256_kernel<<<(Mtok + 7) / 8, 256, 0, stream>>>(x2, ln2_w, ln2_b, h2a, Mtok);

  // 10) MLP: fc1 + exact GELU -> h2b (f16); fc2 + residual x2 -> out (f32)
  gemm128_kernel<1,0><<<dim3((Mtok+127)/128, 8), 256, 0, stream>>>(
      h2a, 256, Mtok, fc1_w, fc1_b, 1024, 256, h2b, 1024, 0, nullptr, 0);
  gemm128_kernel<0,1><<<dim3((Mtok+127)/128, 2), 256, 0, stream>>>(
      h2b, 1024, Mtok, fc2_w, fc2_b, 256, 1024, out, 256, 0, x2, 256);
}
